// TGru_26508538151547
// MI455X (gfx1250) — compile-verified, matrix operand order checked by reading, not run on hardware
//
#include <hip/hip_runtime.h>
#include <math.h>

typedef __attribute__((ext_vector_type(16))) _Float16 v16h;
typedef __attribute__((ext_vector_type(4)))  _Float16 v4h;
typedef __attribute__((ext_vector_type(8)))  float    v8f;

#define IN_F 128
#define HID  64
#define GC   192   // 3 * HID (z|r|h concatenated)
#define FRH  512   // halves per WMMA fragment plane (32 lanes x 16 halves)

__device__ __forceinline__ float sigmoidf_(float x) {
    return 1.0f / (1.0f + __expf(-x));
}

// A/B fragment K mapping (CDNA5 ISA 7.12.2, 16-bit 16x32 A):
// element i of lane (hf = lane>>4):  k = kb*32 + hf*8 + (i&7) + ((i&8) ? 16 : 0)

// ---------------------------------------------------------------------------
// Pre-pack weights into fragment-major f16 layout:
//   Wcp[(nt*4+kb)*512 + lane*16 + i]           nt=0..11 (cols of [Wz|Wr|Wh])
//   Lcp[((g*4+nt)*4+kb)*512 + lane*16 + i]     g=0..2, nt=0..3
// so each lane's B fragment is one contiguous 32-byte load.
// ---------------------------------------------------------------------------
__global__ void k_convert(const float* __restrict__ Wz, const float* __restrict__ Wr,
                          const float* __restrict__ Wh, const float* __restrict__ Lz,
                          const float* __restrict__ Lr, const float* __restrict__ Lh,
                          _Float16* __restrict__ Wcp, _Float16* __restrict__ Lcp) {
    int t = blockIdx.x * blockDim.x + threadIdx.x;
    const int totW = 48 * FRH;   // 12 nt * 4 kb
    if (t < totW) {
        int f = t / FRH, rm = t % FRH;
        int lane = rm >> 4, i = rm & 15;
        int nt = f >> 2, kb = f & 3;
        int hf = lane >> 4;
        int col = nt * 16 + (lane & 15);
        int k = kb * 32 + hf * 8 + (i & 7) + ((i & 8) ? 16 : 0);
        int g = col >> 6, nn = col & 63;
        const float* W = (g == 0) ? Wz : ((g == 1) ? Wr : Wh);
        Wcp[t] = (_Float16)W[k * HID + nn];
    } else if (t < 2 * totW) {
        int u = t - totW;
        int f = u / FRH, rm = u % FRH;
        int lane = rm >> 4, i = rm & 15;
        int g = f >> 4, nt = (f >> 2) & 3, kb = f & 3;
        int hf = lane >> 4;
        int colg = nt * 16 + (lane & 15);
        int k = kb * 32 + hf * 8 + (i & 7) + ((i & 8) ? 16 : 0);
        const float* L = (g == 0) ? Lz : ((g == 1) ? Lr : Lh);
        Lcp[u] = (_Float16)L[k * HID + colg];
    }
}

__global__ void k_deg_init(float* __restrict__ deg, int n) {
    int i = blockIdx.x * blockDim.x + threadIdx.x;
    if (i < n) deg[i] = 1.0f;   // self-loop weight
}

__global__ void k_deg_acc(const int* __restrict__ ei, const float* __restrict__ ew,
                          float* __restrict__ deg, int E) {
    int e = blockIdx.x * blockDim.x + threadIdx.x;
    if (e < E) atomicAdd(&deg[ei[E + e]], ew[e]);   // segment_sum over col (target)
}

__global__ void k_dinv(const float* __restrict__ deg, float* __restrict__ dinv, int n) {
    int i = blockIdx.x * blockDim.x + threadIdx.x;
    if (i < n) {
        float d = deg[i];
        dinv[i] = (d > 0.0f) ? rsqrtf(d) : 0.0f;
    }
}

// Store one 16x16 C tile (wave-uniform fast path when the tile is in-bounds).
__device__ __forceinline__ void store_tile(float* __restrict__ M, v8f acc,
                                           int trow0, int coff, int hf, int n,
                                           bool full) {
    if (full) {
#pragma unroll
        for (int j = 0; j < 8; ++j)
            M[(size_t)(trow0 + j + 8 * hf) * GC + coff] = acc[j];
    } else {
#pragma unroll
        for (int j = 0; j < 8; ++j) {
            int gr = trow0 + j + 8 * hf;
            if (gr < n) M[(size_t)gr * GC + coff] = acc[j];
        }
    }
}

// ---------------------------------------------------------------------------
// M[N,192] = x[N,128] @ [Wz|Wr|Wh]  (f16 WMMA, f32 accumulate)
// Block = 256 threads = 8 waves; block tile = 128 rows; wave tile = 16 rows.
// LDS holds x in A-fragment order; B double-buffered across nt tiles.
// ---------------------------------------------------------------------------
__global__ __launch_bounds__(256) void k_gemm(const float* __restrict__ x,
                                              const _Float16* __restrict__ Wcp,
                                              float* __restrict__ M, int n) {
    __shared__ _Float16 XP[8 * 4 * FRH];   // 8 row-tiles x 4 k-frags, 32 KB
    int tid  = threadIdx.x;
    int row0 = blockIdx.x * 128;
    const float4* x4 = (const float4*)x;
    for (int it = tid; it < 128 * 32; it += 256) {
        int r = it >> 5, c4 = it & 31;
        int gr = row0 + r; if (gr > n - 1) gr = n - 1;
        float4 v = x4[(size_t)gr * 32 + c4];
        int k0 = c4 * 4;
        int tI = r >> 4, m = r & 15;
        int kb = k0 >> 5, rem = k0 & 31;
        int hfk = (rem >> 3) & 1;
        int ib  = (rem & 7) + ((rem & 16) ? 8 : 0);
        int addr = (tI * 4 + kb) * FRH + (hfk * 16 + m) * 16 + ib;
        v4h pv = { (_Float16)v.x, (_Float16)v.y, (_Float16)v.z, (_Float16)v.w };
        *(v4h*)(&XP[addr]) = pv;
    }
    __syncthreads();

    int wave = tid >> 5, lane = tid & 31;
    int hf = lane >> 4, nloc = lane & 15;
    int trow0 = row0 + wave * 16;
    if (trow0 >= n) return;                 // whole wave tile out of range
    bool full = (trow0 + 16 <= n);          // wave-uniform

    v16h A[4];
#pragma unroll
    for (int kb = 0; kb < 4; ++kb)
        A[kb] = *(const v16h*)(&XP[(wave * 4 + kb) * FRH + lane * 16]);

    v16h Ba[4], Bb[4];
#pragma unroll
    for (int kb = 0; kb < 4; ++kb)
        Ba[kb] = *(const v16h*)(Wcp + kb * FRH + lane * 16);

    for (int nt = 0; nt < 12; nt += 2) {
#pragma unroll
        for (int kb = 0; kb < 4; ++kb)
            Bb[kb] = *(const v16h*)(Wcp + ((nt + 1) * 4 + kb) * FRH + lane * 16);
        v8f acc = {};
#pragma unroll
        for (int kb = 0; kb < 4; ++kb)
            acc = __builtin_amdgcn_wmma_f32_16x16x32_f16(
                false, A[kb], false, Ba[kb], (short)0, acc, false, false);
        store_tile(M, acc, trow0, nt * 16 + nloc, hf, n, full);

        int ntn = (nt + 2) % 12;            // wrap-around prefetch (no branch)
#pragma unroll
        for (int kb = 0; kb < 4; ++kb)
            Ba[kb] = *(const v16h*)(Wcp + (ntn * 4 + kb) * FRH + lane * 16);
        v8f acc2 = {};
#pragma unroll
        for (int kb = 0; kb < 4; ++kb)
            acc2 = __builtin_amdgcn_wmma_f32_16x16x32_f16(
                false, A[kb], false, Bb[kb], (short)0, acc2, false, false);
        store_tile(M, acc2, trow0, (nt + 1) * 16 + nloc, hf, n, full);
    }
}

// agg[i,c] = M[i,c] * dinv[i]^2   (self-loop term, also initializes agg)
__global__ void k_selfloop(const float* __restrict__ M, const float* __restrict__ dinv,
                           float* __restrict__ agg, int n) {
    int gid = blockIdx.x * blockDim.x + threadIdx.x;
    if (gid < n * GC) {
        int i = gid / GC;
        float d = dinv[i];
        agg[gid] = M[gid] * d * d;
    }
}

// Edge scatter: agg[col] += M[row] * (dinv[row]*ew*dinv[col]).
// 12 threads per edge, 16 channels each (4x float4 gather + 16 f32 atomics).
// M and agg both fit in the 192MB L2, so atomics stay on-chip.
__global__ void k_edge(const int* __restrict__ ei, const float* __restrict__ ew,
                       const float* __restrict__ dinv, const float* __restrict__ M,
                       float* __restrict__ agg, int E) {
    int gid = blockIdx.x * blockDim.x + threadIdx.x;
    int e = gid / 12;
    if (e >= E) return;
    int chunk = gid % 12;
    int r = ei[e], c = ei[E + e];
    float coeff = dinv[r] * ew[e] * dinv[c];
    const float4* Mr = (const float4*)(M + (size_t)r * GC);
    float* ac = agg + (size_t)c * GC + chunk * 16;
#pragma unroll
    for (int q = 0; q < 4; ++q) {
        float4 v = Mr[chunk * 4 + q];
        atomicAdd(ac + q * 4 + 0, v.x * coeff);
        atomicAdd(ac + q * 4 + 1, v.y * coeff);
        atomicAdd(ac + q * 4 + 2, v.z * coeff);
        atomicAdd(ac + q * 4 + 3, v.w * coeff);
    }
}

// ---------------------------------------------------------------------------
// Fused GRU gates + head. One wave per 16-row tile; per-wave LDS tile holds
// [u_g | h] (and h*R) as f16 in A-fragment order; 16 WMMAs per gate.
// ---------------------------------------------------------------------------
__global__ __launch_bounds__(256) void k_gate(const float* __restrict__ agg,
                                              const float* __restrict__ h,
                                              const float* __restrict__ bz,
                                              const float* __restrict__ br,
                                              const float* __restrict__ bh,
                                              const _Float16* __restrict__ Lcp,
                                              const float* __restrict__ bLz,
                                              const float* __restrict__ bLr,
                                              const float* __restrict__ bLh,
                                              const float* __restrict__ Wo,
                                              const float* __restrict__ bo,
                                              float* __restrict__ out, int n) {
    __shared__ _Float16 TP[8][4 * FRH];   // per-wave 4 k-frags, 32 KB total
    int wave = threadIdx.x >> 5, lane = threadIdx.x & 31;
    int wgid = blockIdx.x * 8 + wave;
    int nwaves = (n + 15) >> 4;
    if (wgid >= nwaves) return;
    int row0 = wgid * 16;
    int hf = lane >> 4, nloc = lane & 15;
    bool full = (row0 + 16 <= n);          // wave-uniform

    // stage h into k = 64..127 (fragment order; 4 consecutive k's stay packed)
    for (int it = lane; it < 16 * 16; it += 32) {
        int r = it >> 4, c4 = it & 15;
        int gr = row0 + r; if (gr > n - 1) gr = n - 1;
        float4 v = *(const float4*)(h + (size_t)gr * HID + c4 * 4);
        int k0 = 64 + c4 * 4;
        int kb = k0 >> 5, rem = k0 & 31;
        int hfk = (rem >> 3) & 1;
        int ib  = (rem & 7) + ((rem & 16) ? 8 : 0);
        int addr = kb * FRH + (hfk * 16 + r) * 16 + ib;
        v4h pv = { (_Float16)v.x, (_Float16)v.y, (_Float16)v.z, (_Float16)v.w };
        *(v4h*)(&TP[wave][addr]) = pv;
    }

    v8f Zv[4], Rv[4], Hv[4], hreg[4];
    const float* bg[3] = {bz, br, bh};
    const float* bL[3] = {bLz, bLr, bLh};

    for (int g = 0; g < 3; ++g) {
        // stage agg_g + bias_g into k = 0..63 (fragment order)
        for (int it = lane; it < 16 * 16; it += 32) {
            int r = it >> 4, c4 = it & 15;
            int gr = row0 + r; if (gr > n - 1) gr = n - 1;
            float4 a4 = *(const float4*)(agg + (size_t)gr * GC + g * HID + c4 * 4);
            float4 b4 = *(const float4*)(bg[g] + c4 * 4);
            int k0 = c4 * 4;
            int kb = k0 >> 5, rem = k0 & 31;
            int hfk = (rem >> 3) & 1;
            int ib  = (rem & 7) + ((rem & 16) ? 8 : 0);
            int addr = kb * FRH + (hfk * 16 + r) * 16 + ib;
            v4h pv = { (_Float16)(a4.x + b4.x), (_Float16)(a4.y + b4.y),
                       (_Float16)(a4.z + b4.z), (_Float16)(a4.w + b4.w) };
            *(v4h*)(&TP[wave][addr]) = pv;
        }
        asm volatile("s_wait_dscnt 0x0" ::: "memory");   // LDS staging complete

        v16h A[4];
#pragma unroll
        for (int kb = 0; kb < 4; ++kb)
            A[kb] = *(const v16h*)(&TP[wave][kb * FRH + lane * 16]);

#pragma unroll
        for (int nt = 0; nt < 4; ++nt) {
            v8f acc = {};
#pragma unroll
            for (int kb = 0; kb < 4; ++kb) {
                v16h B = *(const v16h*)(Lcp + ((g * 4 + nt) * 4 + kb) * FRH + lane * 16);
                acc = __builtin_amdgcn_wmma_f32_16x16x32_f16(
                    false, A[kb], false, B, (short)0, acc, false, false);
            }
            float bias = bL[g][nt * 16 + nloc];
#pragma unroll
            for (int j = 0; j < 8; ++j) acc[j] += bias;
            if (g == 0) {
#pragma unroll
                for (int j = 0; j < 8; ++j) Zv[nt][j] = sigmoidf_(acc[j]);
            } else if (g == 1) {
#pragma unroll
                for (int j = 0; j < 8; ++j) Rv[nt][j] = sigmoidf_(acc[j]);
            } else {
#pragma unroll
                for (int j = 0; j < 8; ++j) Hv[nt][j] = tanhf(acc[j]);
            }
        }

        if (g == 1) {
            // overwrite k = 64..127 with h*R for the candidate gate's A operand
#pragma unroll
            for (int nt = 0; nt < 4; ++nt) {
#pragma unroll
                for (int j = 0; j < 8; ++j) {
                    int mc = j + 8 * hf;
                    int gr = row0 + mc; int grc = (gr > n - 1) ? n - 1 : gr;
                    float hv = h[(size_t)grc * HID + nt * 16 + nloc];
                    hreg[nt][j] = hv;
                    int k = 64 + nt * 16 + nloc;
                    int kb = k >> 5, rem = k & 31;
                    int hfk = (rem >> 3) & 1;
                    int ib  = (rem & 7) + ((rem & 16) ? 8 : 0);
                    TP[wave][kb * FRH + (hfk * 16 + mc) * 16 + ib] =
                        (_Float16)(hv * Rv[nt][j]);
                }
            }
        }
    }

    // Hn = Z*h + (1-Z)*Ht ; head out1 = Hn . Wo + bo
    float po[8];
#pragma unroll
    for (int j = 0; j < 8; ++j) po[j] = 0.0f;
    if (full) {
#pragma unroll
        for (int nt = 0; nt < 4; ++nt) {
            float wo = Wo[nt * 16 + nloc];
#pragma unroll
            for (int j = 0; j < 8; ++j) {
                float z  = Zv[nt][j];
                float hn = z * hreg[nt][j] + (1.0f - z) * Hv[nt][j];
                out[(size_t)n + (size_t)(row0 + j + 8 * hf) * HID + nt * 16 + nloc] = hn;
                po[j] += hn * wo;
            }
        }
    } else {
#pragma unroll
        for (int nt = 0; nt < 4; ++nt) {
            float wo = Wo[nt * 16 + nloc];
#pragma unroll
            for (int j = 0; j < 8; ++j) {
                float z  = Zv[nt][j];
                float hn = z * hreg[nt][j] + (1.0f - z) * Hv[nt][j];
                int gr = row0 + j + 8 * hf;
                if (gr < n) out[(size_t)n + (size_t)gr * HID + nt * 16 + nloc] = hn;
                po[j] += hn * wo;
            }
        }
    }
#pragma unroll
    for (int off = 1; off < 16; off <<= 1)
#pragma unroll
        for (int j = 0; j < 8; ++j) po[j] += __shfl_xor(po[j], off, 32);
    if (nloc == 0) {
        float b0 = bo[0];
        if (full) {
#pragma unroll
            for (int j = 0; j < 8; ++j)
                out[row0 + j + 8 * hf] = po[j] + b0;
        } else {
#pragma unroll
            for (int j = 0; j < 8; ++j) {
                int gr = row0 + j + 8 * hf;
                if (gr < n) out[gr] = po[j] + b0;
            }
        }
    }
}

// ---------------------------------------------------------------------------
extern "C" void kernel_launch(void* const* d_in, const int* in_sizes, int n_in,
                              void* d_out, int out_size, void* d_ws, size_t ws_size,
                              hipStream_t stream) {
    const float* x   = (const float*)d_in[0];
    const int*   ei  = (const int*)d_in[1];
    const float* ew  = (const float*)d_in[2];
    const float* h   = (const float*)d_in[3];
    const float* Wz  = (const float*)d_in[4];  const float* bz  = (const float*)d_in[5];
    const float* Wr  = (const float*)d_in[6];  const float* br  = (const float*)d_in[7];
    const float* Wh  = (const float*)d_in[8];  const float* bh  = (const float*)d_in[9];
    const float* Lz  = (const float*)d_in[10]; const float* bLz = (const float*)d_in[11];
    const float* Lr  = (const float*)d_in[12]; const float* bLr = (const float*)d_in[13];
    const float* Lh  = (const float*)d_in[14]; const float* bLh = (const float*)d_in[15];
    const float* Wo  = (const float*)d_in[16]; const float* bo  = (const float*)d_in[17];
    int n = in_sizes[0] / IN_F;
    int E = in_sizes[2];
    float* out = (float*)d_out;
    (void)n_in; (void)out_size; (void)ws_size;

    char* ws = (char*)d_ws;
    size_t off = 0;
    auto carve = [&](size_t bytes) {
        void* p = ws + off;
        off = (off + bytes + 255) & ~(size_t)255;
        return p;
    };
    float*    M    = (float*)carve((size_t)n * GC * sizeof(float));
    float*    agg  = (float*)carve((size_t)n * GC * sizeof(float));
    float*    deg  = (float*)carve((size_t)n * sizeof(float));
    float*    dinv = (float*)carve((size_t)n * sizeof(float));
    _Float16* Wcp  = (_Float16*)carve((size_t)48 * FRH * sizeof(_Float16));
    _Float16* Lcp  = (_Float16*)carve((size_t)48 * FRH * sizeof(_Float16));

    const int tb = 256;
    k_convert<<<(2 * 48 * FRH + tb - 1) / tb, tb, 0, stream>>>(Wz, Wr, Wh, Lz, Lr, Lh, Wcp, Lcp);
    k_deg_init<<<(n + tb - 1) / tb, tb, 0, stream>>>(deg, n);
    k_deg_acc<<<(E + tb - 1) / tb, tb, 0, stream>>>(ei, ew, deg, E);
    k_dinv<<<(n + tb - 1) / tb, tb, 0, stream>>>(deg, dinv, n);
    k_gemm<<<(n + 127) / 128, 256, 0, stream>>>(x, Wcp, M, n);
    k_selfloop<<<(n * GC + tb - 1) / tb, tb, 0, stream>>>(M, dinv, agg, n);
    k_edge<<<(E * 12 + tb - 1) / tb, tb, 0, stream>>>(ei, ew, dinv, M, agg, E);
    int nwaves = (n + 15) / 16;
    k_gate<<<(nwaves + 7) / 8, 256, 0, stream>>>(agg, h, bz, br, bh, Lcp,
                                                 bLz, bLr, bLh, Wo, bo, out, n);
}
